// MambaLayer_72464688218885
// MI455X (gfx1250) — matrix operand order repaired and checked by reading, not compiled, not run on hardware
//
#include <hip/hip_runtime.h>
#include <hip/hip_bf16.h>

#define DIMC    96
#define D_STATE 16
#define D_INNER 192
#define DT_RANK 6
#define LVOL    13824            // 24*24*24
#define NBATCH  2
#define MROWS   (NBATCH * LVOL)  // 27648
#define LN_EPS  1e-5f
#define CHUNK   128
#define NCH     (LVOL / CHUNK)   // 108

typedef __attribute__((ext_vector_type(16))) _Float16 v16h;
typedef __attribute__((ext_vector_type(8)))  _Float16 v8h;
typedef __attribute__((ext_vector_type(8)))  float    v8f;

// ---------------------------------------------------------------------------
// Kernel 1: pre-swizzle weight matrices into WMMA B-fragment layout (f16).
// Bw layout: [n_tile][k_step][lane(32)][16 halves]; lane holds column
// n = nt*16 + lane%16, k = ks*32 + (lane<16?0:16) + i.
// ---------------------------------------------------------------------------
__global__ void wswizzle_kernel(const float* __restrict__ in_proj_w,   // [384,96]
                                const float* __restrict__ out_proj_w,  // [96,192]
                                _Float16* __restrict__ Bw1,            // 24*3*32*16
                                _Float16* __restrict__ Bw2)            // 6*6*32*16
{
    int t = blockIdx.x * blockDim.x + threadIdx.x;   // grid = 36864 threads exactly
    {
        int i    = t & 15;
        int lane = (t >> 4) & 31;
        int r    = t >> 9;            // nt*3 + ks
        int ks   = r % 3;
        int nt   = r / 3;             // < 24
        int n    = nt * 16 + (lane & 15);
        int k    = ks * 32 + ((lane < 16) ? 0 : 16) + i;
        Bw1[t] = (_Float16)in_proj_w[n * DIMC + k];
    }
    if (t < 6 * 6 * 32 * 16) {
        int i    = t & 15;
        int lane = (t >> 4) & 31;
        int r    = t >> 9;            // nt*6 + ks
        int ks   = r % 6;
        int nt   = r / 6;             // < 6
        int n    = nt * 16 + (lane & 15);
        int k    = ks * 32 + ((lane < 16) ? 0 : 16) + i;
        Bw2[t] = (_Float16)out_proj_w[n * D_INNER + k];
    }
}

// ---------------------------------------------------------------------------
// Kernel 2: LayerNorm over 96 channels, one wave32 per token row, f16 out.
// x is NCDHW: x[(b*96+c)*L + l]; row m = b*L + l.
// ---------------------------------------------------------------------------
__global__ void ln_kernel(const float* __restrict__ x,
                          const float* __restrict__ gamma,
                          const float* __restrict__ beta,
                          _Float16* __restrict__ xn)       // [M, 96]
{
    int wid  = (blockIdx.x * blockDim.x + threadIdx.x) >> 5;  // row m
    int lane = threadIdx.x & 31;
    int b = wid / LVOL, l = wid % LVOL;
    const float* xb = x + (size_t)b * DIMC * LVOL + l;
    float v0 = xb[(size_t)(lane)      * LVOL];
    float v1 = xb[(size_t)(lane + 32) * LVOL];
    float v2 = xb[(size_t)(lane + 64) * LVOL];
    float s1 = v0 + v1 + v2;
    float s2 = v0 * v0 + v1 * v1 + v2 * v2;
    #pragma unroll
    for (int o = 16; o > 0; o >>= 1) {
        s1 += __shfl_xor(s1, o, 32);
        s2 += __shfl_xor(s2, o, 32);
    }
    float mu   = s1 * (1.0f / DIMC);
    float var  = s2 * (1.0f / DIMC) - mu * mu;
    float rstd = rsqrtf(var + LN_EPS);
    _Float16* row = xn + (size_t)wid * DIMC;
    row[lane]      = (_Float16)((v0 - mu) * rstd * gamma[lane]      + beta[lane]);
    row[lane + 32] = (_Float16)((v1 - mu) * rstd * gamma[lane + 32] + beta[lane + 32]);
    row[lane + 64] = (_Float16)((v2 - mu) * rstd * gamma[lane + 64] + beta[lane + 64]);
}

// ---------------------------------------------------------------------------
// WMMA helper: load A fragment (16x32 f16). Lane holds row m=lane%16,
// K-halves {kb..kb+7} and {kb+8..kb+15} -> two 16B loads.
// ---------------------------------------------------------------------------
__device__ inline v16h load_a_frag(const _Float16* row_base, int kbase) {
    v8h a0 = *reinterpret_cast<const v8h*>(row_base + kbase);
    v8h a1 = *reinterpret_cast<const v8h*>(row_base + kbase + 8);
    v16h a;
    #pragma unroll
    for (int i = 0; i < 8; ++i) { a[i] = a0[i]; a[i + 8] = a1[i]; }
    return a;
}

// ---------------------------------------------------------------------------
// Kernel 3: in_proj GEMM, xz[M,384] = xn[M,96] * Bw1.
// One wave per (16-row M tile) x (4 consecutive 16-col N tiles): A fragment
// loaded once per K-step and reused across 4 accumulators. 12 WMMAs/wave.
// Grid: 1728 mtiles * 6 ngroups = 10368 waves = 1296 blocks * 8 waves.
// ---------------------------------------------------------------------------
__global__ void gemm_inproj_kernel(const _Float16* __restrict__ xn,
                                   const _Float16* __restrict__ Bw1,
                                   float* __restrict__ xz)
{
    int wid  = (blockIdx.x * blockDim.x + threadIdx.x) >> 5;
    int lane = threadIdx.x & 31;
    int ng = wid % 6;            // group of 4 N tiles
    int mt = wid / 6;
    int kbase = (lane < 16) ? 0 : 16;
    const _Float16* arow = xn + (size_t)(mt * 16 + (lane & 15)) * DIMC;
    v8f acc[4] = {};
    #pragma unroll
    for (int ks = 0; ks < 3; ++ks) {
        v16h a = load_a_frag(arow + ks * 32, kbase);
        #pragma unroll
        for (int j = 0; j < 4; ++j) {
            int nt = ng * 4 + j;
            v16h bfrag = *reinterpret_cast<const v16h*>(
                Bw1 + (((size_t)nt * 3 + ks) * 32 + lane) * 16);
            acc[j] = __builtin_amdgcn_wmma_f32_16x16x32_f16(
                false, a, false, bfrag, (short)0, acc[j], false, false);
        }
    }
    int mbase = mt * 16 + ((lane >> 4) << 3);
    #pragma unroll
    for (int j = 0; j < 4; ++j) {
        int n = (ng * 4 + j) * 16 + (lane & 15);
        #pragma unroll
        for (int r = 0; r < 8; ++r)
            xz[(size_t)(mbase + r) * 384 + n] = acc[j][r];
    }
}

// ---------------------------------------------------------------------------
// Kernel 4: causal depthwise conv (taps=4) along L within each batch + SiLU.
// Input: x branch = xz[:, 0:192]. Thread per (m, d).
// ---------------------------------------------------------------------------
__global__ void conv_silu_kernel(const float* __restrict__ xz,
                                 const float* __restrict__ conv_w,  // [192,1,4]
                                 const float* __restrict__ conv_b,
                                 float* __restrict__ xc)            // [M,192]
{
    int t = blockIdx.x * blockDim.x + threadIdx.x;   // m*192 + d
    int d = t % D_INNER;
    int m = t / D_INNER;
    int l = m % LVOL;
    float w0 = conv_w[d * 4 + 0], w1 = conv_w[d * 4 + 1];
    float w2 = conv_w[d * 4 + 2], w3 = conv_w[d * 4 + 3];
    float acc = conv_b[d] + w3 * xz[(size_t)m * 384 + d];
    if (l >= 1) acc += w2 * xz[(size_t)(m - 1) * 384 + d];
    if (l >= 2) acc += w1 * xz[(size_t)(m - 2) * 384 + d];
    if (l >= 3) acc += w0 * xz[(size_t)(m - 3) * 384 + d];
    xc[t] = acc / (1.0f + __expf(-acc));             // SiLU
}

// ---------------------------------------------------------------------------
// Kernel 5: x_proj (N=38, VALU dot). Splits outputs into dt_raw / B / C.
// ---------------------------------------------------------------------------
__global__ void xproj_kernel(const float* __restrict__ xc,
                             const float* __restrict__ x_proj_w,   // [38,192]
                             float* __restrict__ dtr,              // [M,6]
                             float* __restrict__ Bbuf,             // [M,16]
                             float* __restrict__ Cbuf)             // [M,16]
{
    int t = blockIdx.x * blockDim.x + threadIdx.x;   // m*38 + o
    int o = t % 38;
    int m = t / 38;
    const float* xr = xc + (size_t)m * D_INNER;
    const float* wr = x_proj_w + (size_t)o * D_INNER;
    float acc = 0.0f;
    #pragma unroll 8
    for (int d = 0; d < D_INNER; ++d) acc += xr[d] * wr[d];
    if (o < DT_RANK)                dtr[m * DT_RANK + o] = acc;
    else if (o < DT_RANK + D_STATE) Bbuf[m * D_STATE + (o - DT_RANK)] = acc;
    else                            Cbuf[m * D_STATE + (o - DT_RANK - D_STATE)] = acc;
}

// ---------------------------------------------------------------------------
// Kernel 6: dt = softplus(dt_raw @ dt_proj_w.T + b). Thread per (m, d).
// ---------------------------------------------------------------------------
__global__ void dt_kernel(const float* __restrict__ dtr,
                          const float* __restrict__ dt_proj_w,   // [192,6]
                          const float* __restrict__ dt_proj_b,
                          float* __restrict__ dt)                // [M,192]
{
    int t = blockIdx.x * blockDim.x + threadIdx.x;   // m*192 + d
    int d = t % D_INNER;
    int m = t / D_INNER;
    const float* r = dtr + (size_t)m * DT_RANK;
    const float* w = dt_proj_w + (size_t)d * DT_RANK;
    float acc = dt_proj_b[d];
    #pragma unroll
    for (int j = 0; j < DT_RANK; ++j) acc += r[j] * w[j];
    dt[t] = (acc > 20.0f) ? acc : log1pf(__expf(acc));
}

// ---------------------------------------------------------------------------
// Chunked parallel scan. Recurrence h' = a*h + b is linear, so each chunk's
// effect is (Aprod, Hend) with h_out = Aprod*h_in + Hend.
// Pass 1: per-(b,chunk,d) local scan from zero state -> chunk transitions.
// Thread map puts consecutive d on consecutive lanes (coalesced dt/xc loads,
// broadcast B rows). 41472 threads.
// ---------------------------------------------------------------------------
__global__ void scan_part1_kernel(const float* __restrict__ dt,
                                  const float* __restrict__ xc,
                                  const float* __restrict__ Bbuf,
                                  const float* __restrict__ A_log,
                                  float* __restrict__ chunkA,   // [B*NCH*192, 16]
                                  float* __restrict__ chunkH)   // [B*NCH*192, 16]
{
    int t = blockIdx.x * blockDim.x + threadIdx.x;   // (b*NCH + ch)*192 + d
    int d  = t % D_INNER;
    int r  = t / D_INNER;
    int ch = r % NCH;
    int b  = r / NCH;
    float Aneg[D_STATE];
    #pragma unroll
    for (int n = 0; n < D_STATE; ++n) Aneg[n] = -__expf(A_log[d * D_STATE + n]);
    float h[D_STATE], ap[D_STATE];
    #pragma unroll
    for (int n = 0; n < D_STATE; ++n) { h[n] = 0.0f; ap[n] = 1.0f; }
    size_t mbase = (size_t)b * LVOL + (size_t)ch * CHUNK;
    for (int i = 0; i < CHUNK; ++i) {
        size_t m   = mbase + i;
        float  dtv = dt[m * D_INNER + d];
        float  du  = dtv * xc[m * D_INNER + d];
        const float* Bp = Bbuf + m * D_STATE;
        #pragma unroll
        for (int n = 0; n < D_STATE; ++n) {
            float a = __expf(dtv * Aneg[n]);
            h[n]  = a * h[n] + du * Bp[n];
            ap[n] *= a;
        }
    }
    float* pa = chunkA + (size_t)t * D_STATE;
    float* ph = chunkH + (size_t)t * D_STATE;
    #pragma unroll
    for (int n = 0; n < D_STATE; ++n) { pa[n] = ap[n]; ph[n] = h[n]; }
}

// ---------------------------------------------------------------------------
// Pass 2: sequential carry across the 108 chunk transitions per (b,d,n).
// carry[ch] = state entering chunk ch. 6144 threads * 108 cheap steps.
// ---------------------------------------------------------------------------
__global__ void scan_part2_kernel(const float* __restrict__ chunkA,
                                  const float* __restrict__ chunkH,
                                  float* __restrict__ carry)
{
    int t = blockIdx.x * blockDim.x + threadIdx.x;   // (b*192 + d)*16 + n
    int n = t % D_STATE;
    int r = t / D_STATE;      // b*192 + d
    int d = r % D_INNER;
    int b = r / D_INNER;
    float c = 0.0f;
    for (int ch = 0; ch < NCH; ++ch) {
        size_t idx = ((((size_t)b * NCH + ch) * D_INNER + d) * D_STATE) + n;
        carry[idx] = c;
        c = chunkA[idx] * c + chunkH[idx];
    }
}

// ---------------------------------------------------------------------------
// Pass 3: re-scan each chunk seeded with its carry; fused with C-reduction,
// D skip connection, SiLU(z) gate, f16 store for out_proj.
// ---------------------------------------------------------------------------
__global__ void scan_part3_kernel(const float* __restrict__ dt,
                                  const float* __restrict__ xc,
                                  const float* __restrict__ Bbuf,
                                  const float* __restrict__ Cbuf,
                                  const float* __restrict__ xz,      // gate z
                                  const float* __restrict__ A_log,
                                  const float* __restrict__ D_param,
                                  const float* __restrict__ carry,
                                  _Float16* __restrict__ y16)        // [M,192]
{
    int t = blockIdx.x * blockDim.x + threadIdx.x;   // (b*NCH + ch)*192 + d
    int d  = t % D_INNER;
    int r  = t / D_INNER;
    int ch = r % NCH;
    int b  = r / NCH;
    float Aneg[D_STATE];
    #pragma unroll
    for (int n = 0; n < D_STATE; ++n) Aneg[n] = -__expf(A_log[d * D_STATE + n]);
    float h[D_STATE];
    const float* cp = carry + (size_t)t * D_STATE;
    #pragma unroll
    for (int n = 0; n < D_STATE; ++n) h[n] = cp[n];
    float Dp = D_param[d];
    size_t mbase = (size_t)b * LVOL + (size_t)ch * CHUNK;
    for (int i = 0; i < CHUNK; ++i) {
        size_t m   = mbase + i;
        float  dtv = dt[m * D_INNER + d];
        float  u   = xc[m * D_INNER + d];
        float  du  = dtv * u;
        const float* Bp = Bbuf + m * D_STATE;
        const float* Cp = Cbuf + m * D_STATE;
        float y = 0.0f;
        #pragma unroll
        for (int n = 0; n < D_STATE; ++n) {
            h[n] = __expf(dtv * Aneg[n]) * h[n] + du * Bp[n];
            y += h[n] * Cp[n];
        }
        y += u * Dp;
        float z = xz[m * 384 + D_INNER + d];
        y *= z / (1.0f + __expf(-z));                // * silu(z)
        y16[m * D_INNER + d] = (_Float16)y;
    }
}

// ---------------------------------------------------------------------------
// Kernel 8: out_proj GEMM with fused transpose store to NCDHW output.
// One wave per 16-row M tile computes the full 96-wide N (6 tiles): A loaded
// once per K-step, reused across 6 accumulators -> 36 WMMAs/wave.
// Grid: 1728 waves = 216 blocks * 8 waves. K loop: 6 steps of 32.
// ---------------------------------------------------------------------------
__global__ void gemm_outproj_kernel(const _Float16* __restrict__ y16,
                                    const _Float16* __restrict__ Bw2,
                                    float* __restrict__ out)
{
    int wid  = (blockIdx.x * blockDim.x + threadIdx.x) >> 5;
    int lane = threadIdx.x & 31;
    int mt = wid;
    int kbase = (lane < 16) ? 0 : 16;
    const _Float16* arow = y16 + (size_t)(mt * 16 + (lane & 15)) * D_INNER;
    v8f acc[6] = {};
    #pragma unroll
    for (int ks = 0; ks < 6; ++ks) {
        v16h a = load_a_frag(arow + ks * 32, kbase);
        #pragma unroll
        for (int nt = 0; nt < 6; ++nt) {
            v16h bfrag = *reinterpret_cast<const v16h*>(
                Bw2 + (((size_t)nt * 6 + ks) * 32 + lane) * 16);
            acc[nt] = __builtin_amdgcn_wmma_f32_16x16x32_f16(
                false, a, false, bfrag, (short)0, acc[nt], false, false);
        }
    }
    int mbase = mt * 16 + ((lane >> 4) << 3);
    #pragma unroll
    for (int nt = 0; nt < 6; ++nt) {
        int c = nt * 16 + (lane & 15);
        #pragma unroll
        for (int j = 0; j < 8; ++j) {
            int m = mbase + j;
            int b = m / LVOL, l = m % LVOL;
            out[((size_t)b * DIMC + c) * LVOL + l] = acc[nt][j];
        }
    }
}

// ---------------------------------------------------------------------------
// Host launcher
// ---------------------------------------------------------------------------
extern "C" void kernel_launch(void* const* d_in, const int* in_sizes, int n_in,
                              void* d_out, int out_size, void* d_ws, size_t ws_size,
                              hipStream_t stream) {
    const float* x          = (const float*)d_in[0];
    const float* gamma      = (const float*)d_in[1];
    const float* beta       = (const float*)d_in[2];
    const float* in_proj_w  = (const float*)d_in[3];
    const float* conv_w     = (const float*)d_in[4];
    const float* conv_b     = (const float*)d_in[5];
    const float* x_proj_w   = (const float*)d_in[6];
    const float* dt_proj_w  = (const float*)d_in[7];
    const float* dt_proj_b  = (const float*)d_in[8];
    const float* A_log      = (const float*)d_in[9];
    const float* D_param    = (const float*)d_in[10];
    const float* out_proj_w = (const float*)d_in[11];
    float* out = (float*)d_out;

    char*  ws  = (char*)d_ws;
    size_t off = 0;
    auto carve = [&](size_t bytes) -> void* {
        void* p = ws + off;
        off = (off + bytes + 255) & ~(size_t)255;
        return p;
    };
    _Float16* xn     = (_Float16*)carve((size_t)MROWS * DIMC * 2);
    float*    xz     = (float*)   carve((size_t)MROWS * 384 * 4);
    float*    xc     = (float*)   carve((size_t)MROWS * D_INNER * 4);
    float*    dtr    = (float*)   carve((size_t)MROWS * DT_RANK * 4);
    float*    Bbuf   = (float*)   carve((size_t)MROWS * D_STATE * 4);
    float*    Cbuf   = (float*)   carve((size_t)MROWS * D_STATE * 4);
    float*    dtb    = (float*)   carve((size_t)MROWS * D_INNER * 4);
    _Float16* y16    = (_Float16*)carve((size_t)MROWS * D_INNER * 2);
    float*    chunkA = (float*)   carve((size_t)NBATCH * NCH * D_INNER * D_STATE * 4);
    float*    chunkH = (float*)   carve((size_t)NBATCH * NCH * D_INNER * D_STATE * 4);
    float*    carry  = (float*)   carve((size_t)NBATCH * NCH * D_INNER * D_STATE * 4);
    _Float16* Bw1    = (_Float16*)carve((size_t)24 * 3 * 32 * 16 * 2);
    _Float16* Bw2    = (_Float16*)carve((size_t)6 * 6 * 32 * 16 * 2);

    // 1) weight pre-swizzle: 36864 threads exactly
    wswizzle_kernel<<<144, 256, 0, stream>>>(in_proj_w, out_proj_w, Bw1, Bw2);
    // 2) layernorm: one wave per row
    ln_kernel<<<MROWS / 8, 256, 0, stream>>>(x, gamma, beta, xn);
    // 3) in_proj GEMM: 1728 mtiles * 6 ngroups waves, 8 waves/block
    gemm_inproj_kernel<<<(MROWS / 16) * 6 / 8, 256, 0, stream>>>(xn, Bw1, xz);
    // 4) causal conv + silu
    conv_silu_kernel<<<(MROWS * D_INNER) / 256, 256, 0, stream>>>(xz, conv_w, conv_b, xc);
    // 5) x_proj: M*38 threads
    xproj_kernel<<<(MROWS * 38) / 256, 256, 0, stream>>>(xc, x_proj_w, dtr, Bbuf, Cbuf);
    // 6) dt
    dt_kernel<<<(MROWS * D_INNER) / 256, 256, 0, stream>>>(dtr, dt_proj_w, dt_proj_b, dtb);
    // 7) chunked parallel scan: 2*108*192 = 41472 threads for passes 1/3
    scan_part1_kernel<<<(NBATCH * NCH * D_INNER) / 256, 256, 0, stream>>>(
        dtb, xc, Bbuf, A_log, chunkA, chunkH);
    scan_part2_kernel<<<(NBATCH * D_INNER * D_STATE) / 256, 256, 0, stream>>>(
        chunkA, chunkH, carry);
    scan_part3_kernel<<<(NBATCH * NCH * D_INNER) / 256, 256, 0, stream>>>(
        dtb, xc, Bbuf, Cbuf, xz, A_log, D_param, carry, y16);
    // 8) out_proj GEMM + transpose store: 1728 waves
    gemm_outproj_kernel<<<(MROWS / 16) / 8, 256, 0, stream>>>(y16, Bw2, out);
}
